// DLRM_67843303408159
// MI455X (gfx1250) — matrix operand order repaired and checked by reading, not compiled, not run on hardware
//
#include <hip/hip_runtime.h>

typedef __attribute__((ext_vector_type(16))) __bf16         v16bf;
typedef __attribute__((ext_vector_type(16))) unsigned short v16u;
typedef __attribute__((ext_vector_type(8)))  unsigned short v8u;
typedef __attribute__((ext_vector_type(4)))  unsigned short v4u;
typedef __attribute__((ext_vector_type(8)))  float          v8f;

__device__ __forceinline__ unsigned short f2bf(float f) {
    unsigned u = __builtin_bit_cast(unsigned, f);
    unsigned r = 0x7FFFu + ((u >> 16) & 1u);          // RNE
    return (unsigned short)((u + r) >> 16);
}
__device__ __forceinline__ float bf2f(unsigned short h) {
    return __builtin_bit_cast(float, (unsigned)h << 16);
}

// ---- CDNA5 async global->LDS copy (16B per lane) + asynccnt waits ---------
// ISA 15.18: the LDS address is the VDST value plus the instruction offset, and the global
// address gets the same INST_OFFSET, so 4 chunks share one base reg pair.
#define ASYNC_CP16(gp, lp, imm)                                               \
    asm volatile("global_load_async_to_lds_b128 %0, %1, off offset:" #imm     \
                 :: "v"((unsigned)(size_t)(lp)),                              \
                    "v"((unsigned long long)(size_t)(gp)) : "memory")

__device__ __forceinline__ void wait_async0() {
    asm volatile("s_wait_asynccnt 0x0" ::: "memory");
}
__device__ __forceinline__ void wait_async4() {
    asm volatile("s_wait_asynccnt 0x4" ::: "memory");
}

__device__ __forceinline__ v16u cat8(v8u lo, v8u hi) {
    return __builtin_shufflevector(lo, hi, 0,1,2,3,4,5,6,7,8,9,10,11,12,13,14,15);
}

// ---------------------------------------------------------------------------
// Conversion / padding kernels (once per call; 2D grids -> no div/mod).
// ---------------------------------------------------------------------------
__global__ __launch_bounds__(256)
void cvt_act32(const float* __restrict__ A,        // M x K f32
               unsigned short* __restrict__ O,     // M x 32 bf16, zero padded
               int K)                              // K <= 32
{
    const int kp = threadIdx.x & 31;
    const size_t m = (size_t)blockIdx.x * 8 + (threadIdx.x >> 5);
    O[m * 32 + kp] = (kp < K) ? f2bf(A[m * K + kp]) : (unsigned short)0;
}

__global__ __launch_bounds__(256)
void cvt_w_transpose(const float* __restrict__ W,     // K x N f32
                     unsigned short* __restrict__ WT, // N x Kp bf16, zero pad
                     int N, int K, int Kp)
{
    // grid: (N/32, Kp/8); reads coalesced along n
    const int n  = blockIdx.x * 32 + (threadIdx.x & 31);
    const int kp = blockIdx.y * 8  + (threadIdx.x >> 5);
    WT[(size_t)n * Kp + kp] = (kp < K) ? f2bf(W[(size_t)kp * N + n]) : (unsigned short)0;
}

// ---------------------------------------------------------------------------
// bf16 WMMA GEMM:  C[M x N](bf16) = relu(A[M x Kp](bf16) * WT[N x Kp]^T + bias)
// Block tile 128x128, BK=32, double-buffered async global->LDS staging.
// 8 waves (4x2); wave = 32x64 = 2x4 tiles of v_wmma_f32_16x16x32_bf16.
// Every fragment = 2x ds_load_b128 (LDS stride 80B, 16B aligned).
// M, N multiples of 128; Kp multiple of 32 (zero padded).
// ---------------------------------------------------------------------------
#define BK 32
#define LSTR 40   // LDS row stride in bf16 elems (80B)

__global__ __launch_bounds__(256)
void gemm_bf16(const unsigned short* __restrict__ A,
               const unsigned short* __restrict__ WT,
               const float* __restrict__ bias,
               unsigned short* __restrict__ C,
               int N, int Kp)
{
    __shared__ __attribute__((aligned(16))) unsigned short As[2][128 * LSTR];
    __shared__ __attribute__((aligned(16))) unsigned short Bs[2][128 * LSTR];

    const int tid  = threadIdx.x;
    const int lane = tid & 31;
    const int wave = tid >> 5;
    const int wrow = wave >> 1;            // 0..3
    const int wcol = wave & 1;             // 0..1
    const int rowBase = blockIdx.y * 128;
    const int colBase = blockIdx.x * 128;
    const int half = lane >> 4;
    const int l16  = lane & 15;

    const int r   = tid & 127;             // staged row within tile
    const int isB = tid >> 7;              // waves 0-3: A rows, 4-7: B rows
    const unsigned short* gbase =
        (isB ? WT + (size_t)(colBase + r) * Kp
             : A  + (size_t)(rowBase + r) * Kp);
    unsigned short* lbase0 = (isB ? &Bs[0][r * LSTR] : &As[0][r * LSTR]);
    unsigned short* lbase1 = (isB ? &Bs[1][r * LSTR] : &As[1][r * LSTR]);

    auto stage = [&](int kt, int buf) {
        const unsigned short* g = gbase + (size_t)kt * BK;
        unsigned short* l = buf ? lbase1 : lbase0;
        ASYNC_CP16(g,      l,      0);
        ASYNC_CP16(g,      l,      16);
        ASYNC_CP16(g,      l,      32);
        ASYNC_CP16(g,      l,      48);
    };

    v8f acc[2][4];
    #pragma unroll
    for (int mt = 0; mt < 2; ++mt)
        #pragma unroll
        for (int nt = 0; nt < 4; ++nt) acc[mt][nt] = {};

    auto compute = [&](int cur) {
        const v8u* pA = (const v8u*)As[cur];
        const v8u* pB = (const v8u*)Bs[cur];
        v16bf afr[2], bfr[4];
        #pragma unroll
        for (int mt = 0; mt < 2; ++mt) {
            const int m = wrow * 32 + mt * 16 + l16;          // LSTR/8 == 5
            v8u lo = pA[m * 5 + half];                        // k = half*8 ..
            v8u hi = pA[m * 5 + 2 + half];                    // k = 16+half*8 ..
            v16u u = cat8(lo, hi);
            afr[mt] = __builtin_bit_cast(v16bf, u);
        }
        #pragma unroll
        for (int nt = 0; nt < 4; ++nt) {
            const int n = wcol * 64 + nt * 16 + l16;
            v8u lo = pB[n * 5 + half];                        // k = half*8 ..
            v8u hi = pB[n * 5 + 2 + half];
            v16u u = cat8(lo, hi);
            bfr[nt] = __builtin_bit_cast(v16bf, u);
        }
        #pragma unroll
        for (int mt = 0; mt < 2; ++mt)
            #pragma unroll
            for (int nt = 0; nt < 4; ++nt)
                acc[mt][nt] = __builtin_amdgcn_wmma_f32_16x16x32_bf16(
                    false, afr[mt], false, bfr[nt], (short)0, acc[mt][nt], false, false);
    };

    const int nk = Kp / BK;
    stage(0, 0);
    // steady state: prefetch kt+1 while computing kt
    for (int kt = 0; kt < nk - 1; ++kt) {
        const int cur = kt & 1;
        stage(kt + 1, cur ^ 1);
        wait_async4();                 // tile kt landed (FIFO order)
        __syncthreads();
        compute(cur);
        __syncthreads();
    }
    // tail
    wait_async0();
    __syncthreads();
    compute((nk - 1) & 1);

    // epilogue: bias + relu -> bf16 ; C/D layout: N = lane&15, M = half*8 + r
    #pragma unroll
    for (int mt = 0; mt < 2; ++mt)
        #pragma unroll
        for (int nt = 0; nt < 4; ++nt) {
            const int col = colBase + wcol * 64 + nt * 16 + l16;
            const float bv = bias[col];
            #pragma unroll
            for (int rr = 0; rr < 8; ++rr) {
                const int row = rowBase + wrow * 32 + mt * 16 + half * 8 + rr;
                float v = acc[mt][nt][rr] + bv;
                v = fmaxf(v, 0.0f);
                C[(size_t)row * N + col] = f2bf(v);
            }
        }
}

// ---------------------------------------------------------------------------
// Fused embedding gather + self-interaction (one wave per sample).
// Z = [fm(26x128); dense(1x128); zeros] bf16 in LDS; S = Z*Z^T via 16 WMMAs;
// pack 351 upper-triangle terms + 128 dense features -> X (bf16, stride 480).
// ---------------------------------------------------------------------------
#define ZSTR 136   // bf16 elems (272B, 16B aligned); ZSTR/8 == 17
#define SSTR 33

__global__ __launch_bounds__(128)
void embed_interact(const float* __restrict__ emb,            // 26 x 100000 x 128 f32
                    const int*   __restrict__ cat_idx,        // B x 26
                    const unsigned short* __restrict__ dense, // B x 128 bf16
                    unsigned short* __restrict__ X)           // B x 480 bf16
{
    __shared__ __attribute__((aligned(16))) unsigned short Z[4][32 * ZSTR];
    __shared__ float S[4][32 * SSTR];

    const int lane = threadIdx.x & 31;
    const int wave = threadIdx.x >> 5;
    const int b    = blockIdx.x * 4 + wave;
    unsigned short* z = Z[wave];
    float*          s = S[wave];

    // gather Z rows; each lane handles 4 contiguous elems (float4 / b64)
    for (int rw = 0; rw < 32; ++rw) {
        v4u o = {0, 0, 0, 0};
        if (rw < 26) {
            const long long idx = (long long)cat_idx[(size_t)b * 26 + rw];
            const float4 f = *(const float4*)(emb + ((size_t)rw * 100000 + idx) * 128 + lane * 4);
            o[0] = f2bf(f.x); o[1] = f2bf(f.y); o[2] = f2bf(f.z); o[3] = f2bf(f.w);
        } else if (rw == 26) {
            o = *(const v4u*)(dense + (size_t)b * 128 + lane * 4);
        }
        *(v4u*)(&z[rw * ZSTR + lane * 4]) = o;
    }
    __syncthreads();

    const int half = lane >> 4;
    const int l16  = lane & 15;
    const v8u* pz = (const v8u*)z;

    v8f acc[2][2];
    #pragma unroll
    for (int mt = 0; mt < 2; ++mt)
        #pragma unroll
        for (int nt = 0; nt < 2; ++nt) acc[mt][nt] = {};

    #pragma unroll
    for (int kt = 0; kt < 4; ++kt) {
        const int kc = kt * 4;                  // k0/8
        v16bf afr[2], bfr[2];
        #pragma unroll
        for (int t = 0; t < 2; ++t) {
            const int row = t * 16 + l16;
            v8u alo = pz[row * 17 + kc + half];
            v8u ahi = pz[row * 17 + kc + 2 + half];
            v16u ua = cat8(alo, ahi);
            afr[t] = __builtin_bit_cast(v16bf, ua);
            v8u blo = pz[row * 17 + kc + half];
            v8u bhi = pz[row * 17 + kc + 2 + half];
            v16u ub = cat8(blo, bhi);
            bfr[t] = __builtin_bit_cast(v16bf, ub);
        }
        #pragma unroll
        for (int mt = 0; mt < 2; ++mt)
            #pragma unroll
            for (int nt = 0; nt < 2; ++nt)
                acc[mt][nt] = __builtin_amdgcn_wmma_f32_16x16x32_bf16(
                    false, afr[mt], false, bfr[nt], (short)0, acc[mt][nt], false, false);
    }

    // spill S (32x32 f32) to LDS
    #pragma unroll
    for (int mt = 0; mt < 2; ++mt)
        #pragma unroll
        for (int nt = 0; nt < 2; ++nt)
            #pragma unroll
            for (int rr = 0; rr < 8; ++rr)
                s[(mt * 16 + half * 8 + rr) * SSTR + nt * 16 + l16] = acc[mt][nt][rr];

    // pack: 351 upper-triangle pairs (i<j of 27) then 128 dense features
    unsigned short* xrow = X + (size_t)b * 480;
    for (int p = lane; p < 351; p += 32) {
        int i = 0, rem = p;
        while (rem >= 26 - i) { rem -= 26 - i; ++i; }
        const int j = i + 1 + rem;
        xrow[p] = f2bf(s[i * SSTR + j]);
    }
    for (int e = lane; e < 128; e += 32)
        xrow[351 + e] = dense[(size_t)b * 128 + e];
    if (lane == 0) xrow[479] = 0;   // pad (WT1 pad row is also zero)
}

// ---------------------------------------------------------------------------
// Final 256 -> 1 + sigmoid: one wave per row, shuffle reduction.
// ---------------------------------------------------------------------------
__global__ __launch_bounds__(256)
void final_sigmoid(const unsigned short* __restrict__ T, // B x 256 bf16
                   const float* __restrict__ Wo,         // 256
                   const float* __restrict__ bo,         // 1
                   float* __restrict__ out)              // B
{
    const int lane = threadIdx.x & 31;
    const int wave = threadIdx.x >> 5;
    const int row  = blockIdx.x * 8 + wave;
    const unsigned short* t = T + (size_t)row * 256;
    float sum = 0.0f;
    #pragma unroll
    for (int e = 0; e < 8; ++e)
        sum += bf2f(t[lane + e * 32]) * Wo[lane + e * 32];
    #pragma unroll
    for (int off = 16; off; off >>= 1)
        sum += __shfl_xor(sum, off, 32);
    if (lane == 0) out[row] = 1.0f / (1.0f + __expf(-(sum + bo[0])));
}

// ---------------------------------------------------------------------------
extern "C" void kernel_launch(void* const* d_in, const int* in_sizes, int n_in,
                              void* d_out, int out_size, void* d_ws, size_t ws_size,
                              hipStream_t stream) {
    const float* dense_x = (const float*)d_in[0];
    const int*   cat_idx = (const int*)  d_in[1];
    const float* emb     = (const float*)d_in[2];
    const float* Wd1 = (const float*)d_in[3];  const float* bd1 = (const float*)d_in[4];
    const float* Wd2 = (const float*)d_in[5];  const float* bd2 = (const float*)d_in[6];
    const float* Wdf = (const float*)d_in[7];  const float* bdf = (const float*)d_in[8];
    const float* Wt1 = (const float*)d_in[9];  const float* bt1 = (const float*)d_in[10];
    const float* Wt2 = (const float*)d_in[11]; const float* bt2 = (const float*)d_in[12];
    const float* Wt3 = (const float*)d_in[13]; const float* bt3 = (const float*)d_in[14];
    const float* Wt4 = (const float*)d_in[15]; const float* bt4 = (const float*)d_in[16];
    const float* Wo  = (const float*)d_in[17]; const float* bo  = (const float*)d_in[18];
    float* out = (float*)d_out;

    const size_t Bsz = 16384;
    unsigned short* p = (unsigned short*)d_ws;
    unsigned short* A0    = p; p += Bsz * 32;        // dense_x bf16, Kp=32
    unsigned short* Wd1T  = p; p += (size_t)512  * 32;
    unsigned short* Wd2T  = p; p += (size_t)256  * 512;
    unsigned short* WdfT  = p; p += (size_t)128  * 256;
    unsigned short* Wt1T  = p; p += (size_t)1024 * 480;
    unsigned short* Wt2T  = p; p += (size_t)1024 * 1024;
    unsigned short* Wt3T  = p; p += (size_t)512  * 1024;
    unsigned short* Wt4T  = p; p += (size_t)256  * 512;
    unsigned short* actA  = p; p += Bsz * 1024;
    unsigned short* actB  = p; p += Bsz * 1024;
    unsigned short* dnsb  = p; p += Bsz * 128;
    unsigned short* xbuf  = p; p += Bsz * 480;

    const dim3 blk(256);
    // ---- per-call precision conversion, zero pad, W transpose (2D grids) ----
    cvt_act32<<<dim3((unsigned)(Bsz / 8)), blk, 0, stream>>>(dense_x, A0, 13);
    cvt_w_transpose<<<dim3(512  / 32, 32  / 8), blk, 0, stream>>>(Wd1, Wd1T, 512,  13,   32);
    cvt_w_transpose<<<dim3(256  / 32, 512 / 8), blk, 0, stream>>>(Wd2, Wd2T, 256,  512,  512);
    cvt_w_transpose<<<dim3(128  / 32, 256 / 8), blk, 0, stream>>>(Wdf, WdfT, 128,  256,  256);
    cvt_w_transpose<<<dim3(1024 / 32, 480 / 8), blk, 0, stream>>>(Wt1, Wt1T, 1024, 479,  480);
    cvt_w_transpose<<<dim3(1024 / 32, 1024/ 8), blk, 0, stream>>>(Wt2, Wt2T, 1024, 1024, 1024);
    cvt_w_transpose<<<dim3(512  / 32, 1024/ 8), blk, 0, stream>>>(Wt3, Wt3T, 512,  1024, 1024);
    cvt_w_transpose<<<dim3(256  / 32, 512 / 8), blk, 0, stream>>>(Wt4, Wt4T, 256,  512,  512);

    // ---- bottom MLP: 13 -> 512 -> 256 -> 128 ----
    gemm_bf16<<<dim3(4, 128), blk, 0, stream>>>(A0,   Wd1T, bd1, actA, 512,  32);
    gemm_bf16<<<dim3(2, 128), blk, 0, stream>>>(actA, Wd2T, bd2, actB, 256,  512);
    gemm_bf16<<<dim3(1, 128), blk, 0, stream>>>(actB, WdfT, bdf, dnsb, 128,  256);
    // ---- embedding gather + pairwise interactions -> X ----
    embed_interact<<<dim3(16384 / 4), dim3(128), 0, stream>>>(emb, cat_idx, dnsb, xbuf);
    // ---- top MLP: 479(480) -> 1024 -> 1024 -> 512 -> 256 ----
    gemm_bf16<<<dim3(8, 128), blk, 0, stream>>>(xbuf, Wt1T, bt1, actA, 1024, 480);
    gemm_bf16<<<dim3(8, 128), blk, 0, stream>>>(actA, Wt2T, bt2, actB, 1024, 1024);
    gemm_bf16<<<dim3(4, 128), blk, 0, stream>>>(actB, Wt3T, bt3, actA, 512,  1024);
    gemm_bf16<<<dim3(2, 128), blk, 0, stream>>>(actA, Wt4T, bt4, actB, 256,  512);
    // ---- 256 -> 1 + sigmoid ----
    final_sigmoid<<<dim3(2048), blk, 0, stream>>>(actB, Wo, bo, out);
}
